// ToFace_38740605010213
// MI455X (gfx1250) — matrix operand order repaired and checked by reading, not compiled, not run on hardware
//
#include <hip/hip_runtime.h>
#include <hip/hip_bf16.h>
#include <stdint.h>

// ---------------------------------------------------------------------------
// Types for CDNA5 WMMA
// ---------------------------------------------------------------------------
typedef __attribute__((ext_vector_type(16))) _Float16 v16h;
typedef __attribute__((ext_vector_type(8)))  float    v8f;

union FragH { uint32_t u[8]; v16h h; };
union AccF  { float    f[8]; v8f  v; };

__constant__ float cDB2_LO[4] = {-0.12940952255092145f, 0.22414386804185735f,
                                  0.836516303737469f,   0.48296291314469025f};
__constant__ float cDB2_HI[4] = {-0.48296291314469025f, 0.836516303737469f,
                                 -0.22414386804185735f, -0.12940952255092145f};
#define INV_SQRT2 0.70710678118654752440f

#define CEILDIV(a,b) (((a)+(b)-1)/(b))

// ---------------------------------------------------------------------------
// Generic WMMA GEMM:  C[M,N] = A[M,Ka] * B[Ka,N] (+bias) (+relu)
// A: f16 row-major panel, Ka multiple of 32, padded K region is zero.
// Bp: prepacked lane-fragment layout (see pack_b).
// mode 0: f32 output [M,ldc], columns n<Nreal only.
// mode 1: f16 panel output [M, Ntiles*16], zeros for n>=Nreal (K-pad for next GEMM).
// ---------------------------------------------------------------------------
__global__ __launch_bounds__(256)
void gemm_wmma(const _Float16* __restrict__ A, const _Float16* __restrict__ Bp,
               float* __restrict__ Cf, _Float16* __restrict__ Ch,
               const float* __restrict__ bias,
               int M, int Ka, int Ntiles, int Nreal, int ldc, int mode, int relu)
{
    const int lane = threadIdx.x & 31;
    const int wv   = threadIdx.x >> 5;
    const int mt   = blockIdx.x * 4 + (wv & 3);
    const int nt   = blockIdx.y * 2 + (wv >> 2);
    if (nt >= Ntiles || mt * 16 >= M) return;   // wave-uniform -> EXEC stays all-ones
    const int kg   = lane >> 4;                 // K-half of the wave
    const int ln16 = lane & 15;
    const int nK   = Ka >> 5;                   // 32-wide K steps
    const int Kd   = Ka >> 1;                   // dwords per A row

    const uint32_t* __restrict__ Au = (const uint32_t*)A;
    const uint32_t* __restrict__ Bu = (const uint32_t*)Bp;
    const uint32_t* arow  = Au + (size_t)(mt * 16 + ln16) * Kd;
    const uint32_t* bbase = Bu + (((size_t)nt * nK) * 32 + lane) * 8;

    AccF c;
#pragma unroll
    for (int r = 0; r < 8; ++r) c.f[r] = 0.0f;

    for (int ks = 0; ks < nK; ++ks) {
        FragH a, b;
        const uint32_t* ap = arow + ks * 16 + 8 * kg;
#pragma unroll
        for (int v = 0; v < 4; ++v) { a.u[v] = ap[v]; a.u[4 + v] = ap[4 + v]; }
        const uint32_t* bp = bbase + (size_t)ks * 256;
#pragma unroll
        for (int v = 0; v < 8; ++v) b.u[v] = bp[v];
        c.v = __builtin_amdgcn_wmma_f32_16x16x32_f16(false, a.h, false, b.h,
                                                     (short)0, c.v, false, false);
    }

    const int   n  = nt * 16 + ln16;
    const float bv = (bias != nullptr && n < Nreal) ? bias[n] : 0.0f;
#pragma unroll
    for (int r = 0; r < 8; ++r) {
        const int m = mt * 16 + r + 8 * kg;
        float val = c.f[r] + bv;
        if (relu) val = fmaxf(val, 0.0f);
        if (mode == 0) {
            if (n < Nreal && m < M) Cf[(size_t)m * ldc + n] = val;
        } else {
            if (m < M) Ch[(size_t)m * (Ntiles * 16) + n] = (_Float16)(n < Nreal ? val : 0.0f);
        }
    }
}

// Pack weight W[N,K] (row-major f32) into B lane-fragment order:
// element idx = ((nt*nK+ks)*32+lane)*8+v  holds dword (B[k0][n], B[k0+1][n]),
// n = nt*16 + lane%16,  k0 = ks*32 + (lane/16)*16 + 2v.  Zero padded.
__global__ void pack_b(const float* __restrict__ W, _Float16* __restrict__ out,
                       int N, int K, int Ka, int Ntiles)
{
    const int nK = Ka >> 5;
    const int total = Ntiles * nK * 256;
    int idx = blockIdx.x * blockDim.x + threadIdx.x;
    if (idx >= total) return;
    int v    = idx & 7;
    int lane = (idx >> 3) & 31;
    int ks   = (idx >> 8) % nK;
    int nt   = (idx >> 8) / nK;
    int n  = nt * 16 + (lane & 15);
    int k0 = ks * 32 + (lane >> 4) * 16 + 2 * v;
    float x0 = (n < N && k0     < K) ? W[(size_t)n * K + k0]     : 0.0f;
    float x1 = (n < N && k0 + 1 < K) ? W[(size_t)n * K + k0 + 1] : 0.0f;
    out[(size_t)idx * 2]     = (_Float16)x0;
    out[(size_t)idx * 2 + 1] = (_Float16)x1;
}

// im2col pack: A[m, k] from NCHW image [128, C, 8, 8]; k = cin*9+ky*3+kx, pad->0
__global__ void pack_im2col(const float* __restrict__ img, _Float16* __restrict__ Ap,
                            int C, int Ka)
{
    int idx = blockIdx.x * blockDim.x + threadIdx.x;
    int total = 8192 * Ka;
    if (idx >= total) return;
    int m = idx / Ka, k = idx - m * Ka;
    float v = 0.0f;
    int K = C * 9;
    if (k < K) {
        int cin = k / 9, r = k - cin * 9, ky = r / 3, kx = r - ky * 3;
        int b = m >> 6, pos = m & 63, y = pos >> 3, x = pos & 7;
        int yy = y + ky - 1, xx = x + kx - 1;
        if ((unsigned)yy < 8u && (unsigned)xx < 8u)
            v = img[((size_t)b * C + cin) * 64 + yy * 8 + xx];
    }
    Ap[idx] = (_Float16)v;
}

// ---------------------------------------------------------------------------
// BatchNorm (train mode, biased stats) helpers. Deterministic reductions.
// ---------------------------------------------------------------------------
__global__ __launch_bounds__(256)
void col_stats(const float* __restrict__ Y, int M, int C, float* __restrict__ mv)
{
    int c = blockIdx.x, tid = threadIdx.x;
    __shared__ float s1[256], s2[256];
    float a = 0.f, b = 0.f;
    for (int m = tid; m < M; m += 256) { float v = Y[(size_t)m * C + c]; a += v; b += v * v; }
    s1[tid] = a; s2[tid] = b; __syncthreads();
    for (int st = 128; st > 0; st >>= 1) {
        if (tid < st) { s1[tid] += s1[tid + st]; s2[tid] += s2[tid + st]; }
        __syncthreads();
    }
    if (tid == 0) { float mean = s1[0] / M; mv[c * 2] = mean; mv[c * 2 + 1] = s2[0] / M - mean * mean; }
}

__global__ void bn_relu_nchw(const float* __restrict__ Y, const float* __restrict__ mv,
                             const float* __restrict__ g, const float* __restrict__ be,
                             float* __restrict__ out, int C)
{
    int idx = blockIdx.x * blockDim.x + threadIdx.x;
    if (idx >= 8192 * C) return;
    int m = idx / C, c = idx - m * C;
    float v = (Y[idx] - mv[2 * c]) * rsqrtf(mv[2 * c + 1] + 1e-5f) * g[c] + be[c];
    v = fmaxf(v, 0.0f);
    int b = m >> 6, pos = m & 63;
    out[((size_t)b * C + c) * 64 + pos] = v;
}

__global__ void ups_transpose(const float* __restrict__ upsn, float* __restrict__ ups_nchw)
{
    int idx = blockIdx.x * blockDim.x + threadIdx.x;
    if (idx >= 8192 * 360) return;
    int m = idx / 360, c = idx - m * 360;
    int b = m >> 6, pos = m & 63;
    ups_nchw[((size_t)b * 360 + c) * 64 + pos] = upsn[idx];
}

// ---------------------------------------------------------------------------
// SWT branch: two circular db2 filters along channel dim, per pixel
// ---------------------------------------------------------------------------
__global__ __launch_bounds__(128)
void swt2_kernel(const float* __restrict__ upsn, float* __restrict__ out)
{
    __shared__ float x[360], c1[360];
    int m = blockIdx.x, b = m >> 6, pos = m & 63;
    const float* row = upsn + (size_t)m * 360;
    for (int i = threadIdx.x; i < 360; i += 128) x[i] = row[i];
    __syncthreads();
    for (int i = threadIdx.x; i < 360; i += 128) {
        float acc = 0.f;
#pragma unroll
        for (int j = 0; j < 4; ++j) { int t = (i + 1 - j + 360) % 360; acc += cDB2_HI[j] * x[t]; }
        c1[i] = (float)INV_SQRT2 * acc;
    }
    __syncthreads();
    float* ob = out + (size_t)b * 46080;
    for (int i = threadIdx.x; i < 360; i += 128) {
        float ca = 0.f, cd = 0.f;
#pragma unroll
        for (int j = 0; j < 4; ++j) {
            int t = (i + 1 - j + 360) % 360;
            ca += cDB2_LO[j] * c1[t]; cd += cDB2_HI[j] * c1[t];
        }
        ob[(size_t)i * 64 + pos]         = (float)INV_SQRT2 * ca;
        ob[(size_t)(360 + i) * 64 + pos] = (float)INV_SQRT2 * cd;
    }
}

// (2,7) conv over the raw-reshaped [8192,1,2,360] rows -> relu -> f16 A panel [32768,384]
__global__ void sconv_kernel(const float* __restrict__ swt2, const float* __restrict__ w,
                             const float* __restrict__ bias, _Float16* __restrict__ As)
{
    int idx = blockIdx.x * blockDim.x + threadIdx.x;
    if (idx >= 32768 * 384) return;
    int j = idx % 384, no = idx / 384, o = no & 3, n = no >> 2;
    float v = 0.0f;
    if (j < 360) {
        int b = n >> 6, t = n & 63;
        const float* g = swt2 + (size_t)b * 46080 + (size_t)t * 720;
        float acc = bias[o];
#pragma unroll
        for (int i = 0; i < 2; ++i)
#pragma unroll
            for (int k = 0; k < 7; ++k) {
                int jj = j - 3 + k;
                if ((unsigned)jj < 360u) acc += w[o * 14 + i * 7 + k] * g[i * 360 + jj];
            }
        v = fmaxf(acc, 0.0f);
    }
    As[idx] = (_Float16)v;
}

__global__ void s3_kernel(const float* __restrict__ s2, const float* __restrict__ w,
                          const float* __restrict__ b, float* __restrict__ depth_out)
{
    int r = blockIdx.x * blockDim.x + threadIdx.x;
    if (r >= 32768) return;
    float acc = b[0];
#pragma unroll
    for (int k = 0; k < 16; ++k) acc += w[k] * s2[(size_t)r * 16 + k];
    depth_out[r] = acc;
}

// du = pos-weight(depth^4) * ups, stored NCHW per-batch flat [B, 360*64]
__global__ void du_kernel(const float* __restrict__ depth, const float* __restrict__ upsn,
                          const float* __restrict__ dw, const float* __restrict__ db,
                          float* __restrict__ du)
{
    int idx = blockIdx.x * blockDim.x + threadIdx.x;
    if (idx >= 8192 * 360) return;
    int m = idx / 360, c = idx - m * 360;
    const float* dp = depth + (size_t)m * 4;
    float pv = db[c];
#pragma unroll
    for (int q = 0; q < 4; ++q) { float t = dp[q]; float t2 = t * t; pv += dw[c * 4 + q] * (t2 * t2); }
    int b = m >> 6, pos = m & 63;
    du[(size_t)b * 23040 + (size_t)c * 64 + pos] = pv * upsn[(size_t)m * 360 + c];
}

// ---------------------------------------------------------------------------
// Intense branch: (1,7) conv recomputed twice (stats pass, normalize pass)
// ---------------------------------------------------------------------------
__global__ __launch_bounds__(256)
void iconv_stats(const float* __restrict__ du, const float* __restrict__ w,
                 const float* __restrict__ bias, float* __restrict__ mv)
{
    const int o = blockIdx.x;
    __shared__ double s1[256], s2[256];
    double a = 0.0, b2 = 0.0;
    const float bo = bias[o];
    for (int t = threadIdx.x; t < 8192 * 360; t += 256) {
        int n = t / 360, j = t - n * 360;
        int bb = n >> 6, tt = n & 63;
        const float* g = du + (size_t)bb * 23040 + (size_t)tt * 360;
        float acc = bo;
#pragma unroll
        for (int k = 0; k < 7; ++k) { int jj = j - 3 + k; if ((unsigned)jj < 360u) acc += w[o * 7 + k] * g[jj]; }
        a += acc; b2 += (double)acc * acc;
    }
    s1[threadIdx.x] = a; s2[threadIdx.x] = b2; __syncthreads();
    for (int st = 128; st > 0; st >>= 1) {
        if (threadIdx.x < st) { s1[threadIdx.x] += s1[threadIdx.x + st]; s2[threadIdx.x] += s2[threadIdx.x + st]; }
        __syncthreads();
    }
    if (threadIdx.x == 0) {
        double cnt = 8192.0 * 360.0, mean = s1[0] / cnt;
        mv[o * 2] = (float)mean; mv[o * 2 + 1] = (float)(s2[0] / cnt - mean * mean);
    }
}

__global__ void iconv_norm(const float* __restrict__ du, const float* __restrict__ w,
                           const float* __restrict__ bias, const float* __restrict__ mv,
                           const float* __restrict__ g, const float* __restrict__ be,
                           _Float16* __restrict__ out)
{
    int idx = blockIdx.x * blockDim.x + threadIdx.x;
    if (idx >= 32768 * 384) return;
    int j = idx % 384, no = idx / 384, o = no & 3, n = no >> 2;
    float v = 0.0f;
    if (j < 360) {
        int bb = n >> 6, tt = n & 63;
        const float* gg = du + (size_t)bb * 23040 + (size_t)tt * 360;
        float acc = bias[o];
#pragma unroll
        for (int k = 0; k < 7; ++k) { int jj = j - 3 + k; if ((unsigned)jj < 360u) acc += w[o * 7 + k] * gg[jj]; }
        acc = (acc - mv[2 * o]) * rsqrtf(mv[2 * o + 1] + 1e-5f) * g[o] + be[o];
        v = fmaxf(acc, 0.0f);
    }
    out[idx] = (_Float16)v;
}

// two fused small linear layers, relu both (N1<=16)
__global__ void mlp2_relu(const float* __restrict__ X, const float* __restrict__ w1,
                          const float* __restrict__ b1, const float* __restrict__ w2,
                          const float* __restrict__ b2, float* __restrict__ out,
                          int R, int K1, int N1, int N2)
{
    int r = blockIdx.x * blockDim.x + threadIdx.x;
    if (r >= R) return;
    const float* x = X + (size_t)r * K1;
    float h[16];
    for (int p = 0; p < N1; ++p) {
        float a = b1[p];
        for (int k = 0; k < K1; ++k) a += w1[p * K1 + k] * x[k];
        h[p] = fmaxf(a, 0.0f);
    }
    float* o = out + (size_t)r * N2;
    for (int p = 0; p < N2; ++p) {
        float a = b2[p];
        for (int k = 0; k < N1; ++k) a += w2[p * N1 + k] * h[k];
        o[p] = fmaxf(a, 0.0f);
    }
}

// ---------------------------------------------------------------------------
// DWT branch
// ---------------------------------------------------------------------------
__global__ void xd_kernel(const float* __restrict__ du, float* __restrict__ xd)
{
    int idx = blockIdx.x * blockDim.x + threadIdx.x;
    if (idx >= 8192 * 180) return;
    int m = idx / 180, i = idx - m * 180;
    int b = m >> 6, pos = m & 63;
    const float* base = du + (size_t)b * 23040 + pos;
    float a = 0.f, d = 0.f;
#pragma unroll
    for (int j = 0; j < 4; ++j) {
        int t = 2 * i + 1 - j;
        if ((unsigned)t < 360u) { float xv = base[(size_t)t * 64]; a += cDB2_LO[j] * xv; d += cDB2_HI[j] * xv; }
    }
    float* ob = xd + (size_t)b * 23040 + pos;
    ob[(size_t)i * 64]         = a;
    ob[(size_t)(180 + i) * 64] = d;
}

__global__ void cdca_kernel(const float* __restrict__ xd, float* __restrict__ cdA,
                            float* __restrict__ caD)
{
    int idx = blockIdx.x * blockDim.x + threadIdx.x;
    if (idx >= 8192 * 90) return;
    int m = idx / 90, i = idx - m * 90;
    int b = m >> 6, pos = m & 63;
    const float* xb = xd + (size_t)b * 23040 + pos;
    float a = 0.f, dd = 0.f;
#pragma unroll
    for (int j = 0; j < 4; ++j) {
        int t = 2 * i + 1 - j;
        if ((unsigned)t < 180u) {
            a  += cDB2_LO[j] * xb[(size_t)(180 + t) * 64];  // approx of D-part
            dd += cDB2_HI[j] * xb[(size_t)t * 64];          // detail of A-part
        }
    }
    cdA[(size_t)b * 5760 + (size_t)i * 64 + pos] = a;
    caD[(size_t)b * 5760 + (size_t)i * 64 + pos] = dd;
}

// full 5-level db1 wavelet packet tree (approx-first order) per 90-row
__global__ void tree_kernel(const float* __restrict__ cdA, const float* __restrict__ caD,
                            float* __restrict__ LfD, float* __restrict__ LfA)
{
    int idx = blockIdx.x * blockDim.x + threadIdx.x;
    if (idx >= 16384) return;
    int path = idx >> 13, n = idx & 8191;
    const float* src = path ? caD : cdA;
    float* dst = path ? LfA : LfD;
    int b = n >> 6, t = n & 63;
    float cur[96], nxt[96];
    const float* sp = src + (size_t)b * 5760 + (size_t)t * 90;
    for (int j = 0; j < 90; ++j) cur[j] = sp[j];
    const int lens[6] = {90, 45, 23, 12, 6, 3};
    int nseg = 1;
    const float s = (float)INV_SQRT2;
    for (int lvl = 0; lvl < 5; ++lvl) {
        int Li = lens[lvl], Lo = lens[lvl + 1];
        for (int seg = 0; seg < nseg; ++seg) {
            const float* in = cur + seg * Li;
            float* oa = nxt + (2 * seg) * Lo;
            float* od = nxt + (2 * seg + 1) * Lo;
            for (int i = 0; i < Lo; ++i) {
                float x0 = in[2 * i];
                float x1 = (2 * i + 1 < Li) ? in[2 * i + 1] : 0.0f;
                oa[i] = s * (x0 + x1);
                od[i] = s * (x0 - x1);
            }
        }
        nseg <<= 1;
        int valid = nseg * Lo;
        for (int j = 0; j < valid; ++j) cur[j] = nxt[j];
    }
    float* dp = dst + (size_t)n * 96;
    for (int j = 0; j < 96; ++j) dp[j] = cur[j];
}

// build z (raw-reshape permutation) fused with (1,3) q-conv -> relu -> f16 panel [32768,64]
__global__ void aq_kernel(const float* __restrict__ LfD, const float* __restrict__ LfA,
                          const float* __restrict__ qcw, const float* __restrict__ qcb,
                          _Float16* __restrict__ Aq)
{
    int idx = blockIdx.x * blockDim.x + threadIdx.x;
    if (idx >= 32768 * 64) return;
    int r = idx & 63, mo = idx >> 6, o = mo & 3, m = mo >> 2;
    int b = m >> 6, rm = m & 63;
    const float* Lf = (r < 32) ? LfD : LfA;
    int u = (r < 32) ? r : r - 32;
    float acc = qcb[o];
#pragma unroll
    for (int v = 0; v < 3; ++v) {
        int e   = u * 3 + v;
        int f   = rm * 96 + e;
        int ch  = f >> 6, pos = f & 63;
        int l   = ch / 3, cp = ch - l * 3;
        int fp  = cp * 64 + pos;
        int t   = fp / 3, j = fp - t * 3;
        float z = Lf[((size_t)(b * 64 + t)) * 96 + l * 3 + j];
        acc += qcw[o * 3 + v] * z;
    }
    Aq[idx] = (_Float16)fmaxf(acc, 0.0f);
}

// o = MLP(it + pw)  -> d_out[0:32768]
__global__ void final_kernel(const float* __restrict__ it, const float* __restrict__ zq,
                             const float* __restrict__ w1, const float* __restrict__ b1,
                             const float* __restrict__ w2, const float* __restrict__ b2,
                             const float* __restrict__ w3, const float* __restrict__ b3,
                             float* __restrict__ o_out)
{
    int idx = blockIdx.x * blockDim.x + threadIdx.x;
    if (idx >= 32768) return;
    int n = idx >> 2, q = idx & 3;
    int b = n >> 6, t = n & 63;
    float vin[8];
#pragma unroll
    for (int a = 0; a < 8; ++a) {
        int f  = q * 512 + a * 64 + t;
        int t2 = f >> 5, rem = f & 31, qq = rem >> 3, e = rem & 7;
        float pw = zq[(((size_t)(b * 64 + t2)) * 4 + qq) * 8 + e];
        vin[a] = it[(size_t)idx * 8 + a] + pw;
    }
    float h1[4];
#pragma unroll
    for (int p = 0; p < 4; ++p) {
        float a = b1[p];
#pragma unroll
        for (int k = 0; k < 8; ++k) a += w1[p * 8 + k] * vin[k];
        h1[p] = fmaxf(a, 0.0f);
    }
    float h2[2];
#pragma unroll
    for (int p = 0; p < 2; ++p) {
        float a = b2[p];
#pragma unroll
        for (int k = 0; k < 4; ++k) a += w2[p * 4 + k] * h1[k];
        h2[p] = fmaxf(a, 0.0f);
    }
    o_out[idx] = b3[0] + w3[0] * h2[0] + w3[1] * h2[1];
}

// ---------------------------------------------------------------------------
// Host-side orchestration (graph-capture safe: only kernel launches on stream)
// Workspace requirement: ~88 MB.
// ---------------------------------------------------------------------------
extern "C" void kernel_launch(void* const* d_in, const int* in_sizes, int n_in,
                              void* d_out, int out_size, void* d_ws, size_t ws_size,
                              hipStream_t stream)
{
    (void)in_sizes; (void)n_in; (void)out_size; (void)ws_size;
    const float* x     = (const float*)d_in[0];
    const float* u_w1  = (const float*)d_in[1];
    const float* u_b1  = (const float*)d_in[2];
    const float* u_g1  = (const float*)d_in[3];
    const float* u_be1 = (const float*)d_in[4];
    const float* u_w2  = (const float*)d_in[5];
    const float* u_b2  = (const float*)d_in[6];
    const float* u_g2  = (const float*)d_in[7];
    const float* u_be2 = (const float*)d_in[8];
    const float* u_w3  = (const float*)d_in[9];
    const float* u_b3  = (const float*)d_in[10];
    const float* s_cw  = (const float*)d_in[11];
    const float* s_cb  = (const float*)d_in[12];
    const float* s_w1  = (const float*)d_in[13];
    const float* s_b1  = (const float*)d_in[14];
    const float* s_w2  = (const float*)d_in[15];
    const float* s_b2  = (const float*)d_in[16];
    const float* s_w3  = (const float*)d_in[17];
    const float* s_b3  = (const float*)d_in[18];
    const float* d_w   = (const float*)d_in[19];
    const float* d_b   = (const float*)d_in[20];
    const float* i_cw  = (const float*)d_in[21];
    const float* i_cb  = (const float*)d_in[22];
    const float* i_g   = (const float*)d_in[23];
    const float* i_be  = (const float*)d_in[24];
    const float* i_w1  = (const float*)d_in[25];
    const float* i_b1  = (const float*)d_in[26];
    const float* i_w2  = (const float*)d_in[27];
    const float* i_b2  = (const float*)d_in[28];
    const float* i_w3  = (const float*)d_in[29];
    const float* i_b3  = (const float*)d_in[30];
    const float* i_w4  = (const float*)d_in[31];
    const float* i_b4  = (const float*)d_in[32];
    const float* q_cw  = (const float*)d_in[33];
    const float* q_cb  = (const float*)d_in[34];
    const float* q_w1  = (const float*)d_in[35];
    const float* q_b1  = (const float*)d_in[36];
    const float* q_w2  = (const float*)d_in[37];
    const float* q_b2  = (const float*)d_in[38];
    const float* q_w3  = (const float*)d_in[39];
    const float* q_b3  = (const float*)d_in[40];
    const float* o_w1  = (const float*)d_in[41];
    const float* o_b1  = (const float*)d_in[42];
    const float* o_w2  = (const float*)d_in[43];
    const float* o_b2  = (const float*)d_in[44];
    const float* o_w3  = (const float*)d_in[45];
    const float* o_b3  = (const float*)d_in[46];

    float* out = (float*)d_out;              // [o 32768 | depth 32768 | ups 2949120]
    char*  ws  = (char*)d_ws;

    // ---- workspace layout (bytes), with phase-based reuse of the big arena ----
    const size_t PB1 = 0x000000, PB2 = 0x080000, PB3 = 0x100000;
    const size_t PBS1 = 0x200000, PBI1 = 0x220000, PBS2 = 0x240000;
    const size_t PBI2 = 0x250000, PBQ1 = 0x260000, STAT = 0x270000;
    const size_t ITB = 0x300000, ZQB = 0x400000, DUB = 0x500000, UPSN = 0x1100000;
    const size_t AR = 0x2000000;
    const size_t A1 = AR, Y1 = AR + 0x400000, H1B = AR + 0x600000, A2 = AR + 0x800000;
    const size_t Y2 = AR + 0x1200000, H2B = AR + 0x1600000, A3 = AR + 0x1A00000;
    const size_t SWT2 = AR, ASB = AR + 0x1800000, S1P = AR + 0x3000000, S2B = AR + 0x3600000;
    const size_t AI16 = AR, I1P = AR + 0x1800000, I2B = AR + 0x1E00000;
    const size_t XDB = AR, CDA = AR + 0xC00000, CAD = AR + 0xF00000;
    const size_t LFD = AR + 0x1200000, LFA = AR + 0x1500000, AQB = AR + 0x1800000;
    const size_t Q1B = AR + 0x1C00000;

    #define Fp(off) ((float*)(ws + (off)))
    #define Hp(off) ((_Float16*)(ws + (off)))
    const int T = 256;

    // ---- weight packs (f16 lane-fragment B panels) ----
    pack_b<<<CEILDIV(4 * 8 * 256, T),  T, 0, stream>>>(u_w1, Hp(PB1), 64, 252, 256, 4);
    pack_b<<<CEILDIV(8 * 18 * 256, T), T, 0, stream>>>(u_w2, Hp(PB2), 128, 576, 576, 8);
    pack_b<<<CEILDIV(23 * 36 * 256, T),T, 0, stream>>>(u_w3, Hp(PB3), 360, 1152, 1152, 23);
    pack_b<<<CEILDIV(6 * 12 * 256, T), T, 0, stream>>>(s_w1, Hp(PBS1), 90, 360, 384, 6);
    pack_b<<<CEILDIV(6 * 12 * 256, T), T, 0, stream>>>(i_w1, Hp(PBI1), 90, 360, 384, 6);
    pack_b<<<CEILDIV(1 * 3 * 256, T),  T, 0, stream>>>(s_w2, Hp(PBS2), 16, 90, 96, 1);
    pack_b<<<CEILDIV(2 * 3 * 256, T),  T, 0, stream>>>(i_w2, Hp(PBI2), 32, 90, 96, 2);
    pack_b<<<CEILDIV(2 * 2 * 256, T),  T, 0, stream>>>(q_w1, Hp(PBQ1), 32, 64, 64, 2);

    // ---- Upsample1: conv->BN->relu x2, conv3 -> ups ----
    pack_im2col<<<CEILDIV(8192 * 256, T), T, 0, stream>>>(x, Hp(A1), 28, 256);
    gemm_wmma<<<dim3(128, 2), 256, 0, stream>>>(Hp(A1), Hp(PB1), Fp(Y1), nullptr, u_b1,
                                                8192, 256, 4, 64, 64, 0, 0);
    col_stats<<<64, 256, 0, stream>>>(Fp(Y1), 8192, 64, Fp(STAT));
    bn_relu_nchw<<<CEILDIV(8192 * 64, T), T, 0, stream>>>(Fp(Y1), Fp(STAT), u_g1, u_be1, Fp(H1B), 64);

    pack_im2col<<<CEILDIV(8192 * 576, T), T, 0, stream>>>(Fp(H1B), Hp(A2), 64, 576);
    gemm_wmma<<<dim3(128, 4), 256, 0, stream>>>(Hp(A2), Hp(PB2), Fp(Y2), nullptr, u_b2,
                                                8192, 576, 8, 128, 128, 0, 0);
    col_stats<<<128, 256, 0, stream>>>(Fp(Y2), 8192, 128, Fp(STAT + 0x1000));
    bn_relu_nchw<<<CEILDIV(8192 * 128, T), T, 0, stream>>>(Fp(Y2), Fp(STAT + 0x1000), u_g2, u_be2, Fp(H2B), 128);

    pack_im2col<<<CEILDIV(8192 * 1152, T), T, 0, stream>>>(Fp(H2B), Hp(A3), 128, 1152);
    gemm_wmma<<<dim3(128, 12), 256, 0, stream>>>(Hp(A3), Hp(PB3), Fp(UPSN), nullptr, u_b3,
                                                 8192, 1152, 23, 360, 360, 0, 0);
    ups_transpose<<<CEILDIV(8192 * 360, T), T, 0, stream>>>(Fp(UPSN), out + 65536);

    // ---- SWT branch -> depth ----
    swt2_kernel<<<8192, 128, 0, stream>>>(Fp(UPSN), Fp(SWT2));
    sconv_kernel<<<CEILDIV(32768 * 384, T), T, 0, stream>>>(Fp(SWT2), s_cw, s_cb, Hp(ASB));
    gemm_wmma<<<dim3(512, 3), 256, 0, stream>>>(Hp(ASB), Hp(PBS1), nullptr, Hp(S1P), s_b1,
                                                32768, 384, 6, 90, 0, 1, 1);
    gemm_wmma<<<dim3(512, 1), 256, 0, stream>>>(Hp(S1P), Hp(PBS2), Fp(S2B), nullptr, s_b2,
                                                32768, 96, 1, 16, 16, 0, 1);
    s3_kernel<<<CEILDIV(32768, T), T, 0, stream>>>(Fp(S2B), s_w3, s_b3, out + 32768);
    du_kernel<<<CEILDIV(8192 * 360, T), T, 0, stream>>>(out + 32768, Fp(UPSN), d_w, d_b, Fp(DUB));

    // ---- Intense branch -> it ----
    iconv_stats<<<4, 256, 0, stream>>>(Fp(DUB), i_cw, i_cb, Fp(STAT + 0x2000));
    iconv_norm<<<CEILDIV(32768 * 384, T), T, 0, stream>>>(Fp(DUB), i_cw, i_cb, Fp(STAT + 0x2000),
                                                          i_g, i_be, Hp(AI16));
    gemm_wmma<<<dim3(512, 3), 256, 0, stream>>>(Hp(AI16), Hp(PBI1), nullptr, Hp(I1P), i_b1,
                                                32768, 384, 6, 90, 0, 1, 1);
    gemm_wmma<<<dim3(512, 1), 256, 0, stream>>>(Hp(I1P), Hp(PBI2), Fp(I2B), nullptr, i_b2,
                                                32768, 96, 2, 32, 32, 0, 1);
    mlp2_relu<<<CEILDIV(32768, T), T, 0, stream>>>(Fp(I2B), i_w3, i_b3, i_w4, i_b4, Fp(ITB),
                                                   32768, 32, 16, 8);

    // ---- DWT branch -> zq ----
    xd_kernel<<<CEILDIV(8192 * 180, T), T, 0, stream>>>(Fp(DUB), Fp(XDB));
    cdca_kernel<<<CEILDIV(8192 * 90, T), T, 0, stream>>>(Fp(XDB), Fp(CDA), Fp(CAD));
    tree_kernel<<<CEILDIV(16384, T), T, 0, stream>>>(Fp(CDA), Fp(CAD), Fp(LFD), Fp(LFA));
    aq_kernel<<<CEILDIV(32768 * 64, T), T, 0, stream>>>(Fp(LFD), Fp(LFA), q_cw, q_cb, Hp(AQB));
    gemm_wmma<<<dim3(512, 1), 256, 0, stream>>>(Hp(AQB), Hp(PBQ1), Fp(Q1B), nullptr, q_b1,
                                                32768, 64, 2, 32, 32, 0, 1);
    mlp2_relu<<<CEILDIV(32768, T), T, 0, stream>>>(Fp(Q1B), q_w2, q_b2, q_w3, q_b3, Fp(ZQB),
                                                   32768, 32, 16, 8);

    // ---- output head ----
    final_kernel<<<CEILDIV(32768, T), T, 0, stream>>>(Fp(ITB), Fp(ZQB),
                                                      o_w1, o_b1, o_w2, o_b2, o_w3, o_b3, out);
}
